// MixLinear_GEMM_78065325572611
// MI455X (gfx1250) — 16-bit and 8-bit matrix operand order repaired and checked by reading, compiled, not run on hardware
//
#include <hip/hip_runtime.h>
#include <stdint.h>

#define M_DIM 8192
#define K_DIM 4096
#define N_DIM 4096
#define NFP   256

typedef __attribute__((ext_vector_type(2)))  int      v2i;
typedef __attribute__((ext_vector_type(4)))  int      v4i;
typedef __attribute__((ext_vector_type(8)))  int      v8i;
typedef __attribute__((ext_vector_type(8)))  float    v8f;
typedef __attribute__((ext_vector_type(16))) _Float16 v16h;

union H16 { v4i i[2]; v16h h; };

#if __has_builtin(__builtin_amdgcn_global_load_async_to_lds_b128)
#define HAVE_ASYNC_LDS 1
typedef __attribute__((address_space(1))) v4i gv4i_t;
typedef __attribute__((address_space(3))) v4i lv4i_t;
__device__ __forceinline__ void async_b128(const void* g, void* l) {
    __builtin_amdgcn_global_load_async_to_lds_b128((gv4i_t*)g, (lv4i_t*)l, 0, 0);
}
#else
#define HAVE_ASYNC_LDS 0
__device__ __forceinline__ void async_b128(const void* g, void* l) {
    *(v4i*)l = *(const v4i*)g;
}
#endif

#if HAVE_ASYNC_LDS
#if __has_builtin(__builtin_amdgcn_s_wait_asynccnt)
#define WAIT_ASYNC(n) __builtin_amdgcn_s_wait_asynccnt(n)
#else
#define WAIT_ASYNC(n) asm volatile("s_wait_asynccnt %0" ::"n"(n) : "memory")
#endif
#else
#define WAIT_ASYNC(n) ((void)0)
#endif

__global__ __launch_bounds__(256) void k_quant(const float* __restrict__ x,
                                               int8_t* __restrict__ qx,
                                               _Float16* __restrict__ act,
                                               float* __restrict__ xscale)
{
    const int row = blockIdx.x;
    const int t   = threadIdx.x;
    const float4* xr = (const float4*)(x + (size_t)row * K_DIM);

    float4 v[4];
    float mx = 0.0f;
#pragma unroll
    for (int i = 0; i < 4; ++i) {
        const int f = t + i * 256;
        v[i] = xr[f];
        if ((f & 3) == 0) {
            act[(size_t)row * NFP + (f >> 2)] = (_Float16)v[i].x;
            v[i].x = 0.0f;
        }
        mx = fmaxf(mx, fabsf(v[i].x));
        mx = fmaxf(mx, fabsf(v[i].y));
        mx = fmaxf(mx, fabsf(v[i].z));
        mx = fmaxf(mx, fabsf(v[i].w));
    }

    __shared__ float red[256];
    red[t] = mx;
    __syncthreads();
#pragma unroll
    for (int s = 128; s > 0; s >>= 1) {
        if (t < s) red[t] = fmaxf(red[t], red[t + s]);
        __syncthreads();
    }
    const float amax  = fmaxf(red[0], 1e-30f);
    const float scale = amax * (1.0f / 127.0f);
    const float inv   = 127.0f / amax;
    if (t == 0) xscale[row] = scale;

    int* qrow = (int*)(qx + (size_t)row * K_DIM);
#pragma unroll
    for (int i = 0; i < 4; ++i) {
        const int f = t + i * 256;
        int b0 = (int)rintf(v[i].x * inv);
        int b1 = (int)rintf(v[i].y * inv);
        int b2 = (int)rintf(v[i].z * inv);
        int b3 = (int)rintf(v[i].w * inv);
        b0 = b0 < -128 ? -128 : (b0 > 127 ? 127 : b0);
        b1 = b1 < -128 ? -128 : (b1 > 127 ? 127 : b1);
        b2 = b2 < -128 ? -128 : (b2 > 127 ? 127 : b2);
        b3 = b3 < -128 ? -128 : (b3 > 127 ? 127 : b3);
        qrow[f] = (b0 & 255) | ((b1 & 255) << 8) | ((b2 & 255) << 16) | ((b3 & 255) << 24);
    }
}

__global__ __launch_bounds__(256) void k_wcache(const int8_t* __restrict__ qw,
                                                const int* __restrict__ ind,
                                                _Float16* __restrict__ w16)
{
    const int n = blockIdx.x;
    const int j = threadIdx.x;
    w16[(size_t)n * NFP + j] = (_Float16)(float)qw[(size_t)n * K_DIM + ind[j]];
}

#define BUFSZ 16384

__global__ __launch_bounds__(256) void k_gemm(const int8_t* __restrict__ qx,
                                              const int8_t* __restrict__ qw,
                                              const _Float16* __restrict__ act,
                                              const _Float16* __restrict__ w16,
                                              const float* __restrict__ xscale,
                                              const float* __restrict__ scale_col,
                                              const float* __restrict__ bias,
                                              float* __restrict__ out)
{
    __shared__ __align__(16) uint8_t smem[2 * BUFSZ];

    const int t     = threadIdx.x;
    const int lane  = t & 31;
    const int wave  = t >> 5;
    const int waveM = wave & 3;
    const int waveN = wave >> 2;
    const int blockM = blockIdx.y * 128;
    const int blockN = blockIdx.x * 128;
    const int lrow = lane & 15;
    const int hi   = lane >> 4;

    v8i iacc[2][4] = {};
    v8f facc[2][4] = {};

    const int ldRow = t >> 1;
    const int ldOff = (t & 1) * 32;

    {
        const int8_t* gA = qx + (size_t)(blockM + ldRow) * K_DIM + ldOff;
        const int8_t* gB = qw + (size_t)(blockN + ldRow) * K_DIM + ldOff;
        uint8_t* dRow0 = smem + ldRow * 64 + ldOff;

        async_b128(gA,      dRow0);
        async_b128(gA + 16, dRow0 + 16);
        async_b128(gB,      dRow0 + 8192);
        async_b128(gB + 16, dRow0 + 8192 + 16);

        const int NCHUNK = K_DIM / 64;
        for (int kc = 0; kc < NCHUNK; ++kc) {
            if (kc + 1 < NCHUNK) {
                const int k1 = (kc + 1) * 64;
                uint8_t* d = dRow0 + ((kc + 1) & 1) * BUFSZ;
                async_b128(gA + k1,      d);
                async_b128(gA + k1 + 16, d + 16);
                async_b128(gB + k1,      d + 8192);
                async_b128(gB + k1 + 16, d + 8192 + 16);
                WAIT_ASYNC(4);
            } else {
                WAIT_ASYNC(0);
            }
            __syncthreads();

            const uint8_t* sA = smem + (kc & 1) * BUFSZ;
            const uint8_t* sB = sA + 8192;

            v8i afrag[2];
#pragma unroll
            for (int tm = 0; tm < 2; ++tm) {
                const uint8_t* pa = sA + (waveM * 32 + tm * 16 + lrow) * 64 + hi * 16;
                v2i c0 = *(const v2i*)(pa);
                v2i c1 = *(const v2i*)(pa + 8);
                v2i c2 = *(const v2i*)(pa + 32);
                v2i c3 = *(const v2i*)(pa + 40);
                v8i a = {c0.x, c0.y, c1.x, c1.y, c2.x, c2.y, c3.x, c3.y};
                afrag[tm] = a;
            }
#pragma unroll
            for (int tn = 0; tn < 4; ++tn) {
                const uint8_t* pb = sB + (waveN * 64 + tn * 16 + lrow) * 64 + hi * 16;
                v4i d0 = *(const v4i*)(pb);
                v4i d1 = *(const v4i*)(pb + 32);
                v8i bfrag = {d0.x, d0.y, d0.z, d0.w, d1.x, d1.y, d1.z, d1.w};
#pragma unroll
                for (int tm = 0; tm < 2; ++tm)
                    iacc[tm][tn] = __builtin_amdgcn_wmma_i32_16x16x64_iu8(
                        true, afrag[tm], true, bfrag, iacc[tm][tn], false, false);
            }
            __syncthreads();
        }
    }

    {
        const uint8_t* gA = (const uint8_t*)act + (size_t)(blockM + ldRow) * (NFP * 2) + ldOff;
        const uint8_t* gB = (const uint8_t*)w16 + (size_t)(blockN + ldRow) * (NFP * 2) + ldOff;
        uint8_t* dRow0 = smem + ldRow * 64 + ldOff;

        async_b128(gA,      dRow0);
        async_b128(gA + 16, dRow0 + 16);
        async_b128(gB,      dRow0 + 8192);
        async_b128(gB + 16, dRow0 + 8192 + 16);

        const int NCHUNK = NFP / 32;
        for (int kc = 0; kc < NCHUNK; ++kc) {
            if (kc + 1 < NCHUNK) {
                const int k1 = (kc + 1) * 64;
                uint8_t* d = dRow0 + ((kc + 1) & 1) * BUFSZ;
                async_b128(gA + k1,      d);
                async_b128(gA + k1 + 16, d + 16);
                async_b128(gB + k1,      d + 8192);
                async_b128(gB + k1 + 16, d + 8192 + 16);
                WAIT_ASYNC(4);
            } else {
                WAIT_ASYNC(0);
            }
            __syncthreads();

            const _Float16* hA = (const _Float16*)(smem + (kc & 1) * BUFSZ);
            const _Float16* hB = hA + 4096;

            H16 a[2];
#pragma unroll
            for (int tm = 0; tm < 2; ++tm) {
                const _Float16* pa = hA + (waveM * 32 + tm * 16 + lrow) * 32 + hi * 8;
                a[tm].i[0] = *(const v4i*)(pa);
                a[tm].i[1] = *(const v4i*)(pa + 16);
            }
#pragma unroll
            for (int tn = 0; tn < 4; ++tn) {
                const _Float16* pb = hB + (waveN * 64 + tn * 16 + lrow) * 32 + hi * 8;
                H16 b;
                b.i[0] = *(const v4i*)(pb);
                b.i[1] = *(const v4i*)(pb + 16);
#pragma unroll
                for (int tm = 0; tm < 2; ++tm)
                    facc[tm][tn] = __builtin_amdgcn_wmma_f32_16x16x32_f16(
                        false, a[tm].h, false, b.h, (short)0, facc[tm][tn], false, false);
            }
            __syncthreads();
        }
    }

#pragma unroll
    for (int tm = 0; tm < 2; ++tm) {
        const int rowbase = blockM + waveM * 32 + tm * 16 + hi * 8;
#pragma unroll
        for (int r = 0; r < 8; ++r) {
            const int row = rowbase + r;
            const float xs = xscale[row];
#pragma unroll
            for (int tn = 0; tn < 4; ++tn) {
                const int col = blockN + waveN * 64 + tn * 16 + lrow;
                const float y = scale_col[col] *
                                ((float)iacc[tm][tn][r] * xs + facc[tm][tn][r]) +
                                bias[col];
                out[(size_t)row * N_DIM + col] = y;
            }
        }
    }
}

extern "C" void kernel_launch(void* const* d_in, const int* in_sizes, int n_in,
                              void* d_out, int out_size, void* d_ws, size_t ws_size,
                              hipStream_t stream)
{
    (void)in_sizes; (void)n_in; (void)out_size; (void)ws_size;

    const float*  x         = (const float*)d_in[0];
    const int8_t* q_weight  = (const int8_t*)d_in[1];
    const float*  scale_col = (const float*)d_in[2];
    const float*  bias      = (const float*)d_in[3];
    const int*    ind       = (const int*)d_in[4];
    float* out = (float*)d_out;

    uint8_t* ws = (uint8_t*)d_ws;
    size_t off = 0;
    int8_t*   qx  = (int8_t*)(ws + off);          off += (size_t)M_DIM * K_DIM;
    _Float16* act = (_Float16*)(ws + off);        off += (size_t)M_DIM * NFP * 2;
    _Float16* w16 = (_Float16*)(ws + off);        off += (size_t)N_DIM * NFP * 2;
    float*    xs  = (float*)(ws + off);

    k_quant <<<M_DIM, 256, 0, stream>>>(x, qx, act, xs);
    k_wcache<<<N_DIM, 256, 0, stream>>>(q_weight, ind, w16);

    dim3 grid(N_DIM / 128, M_DIM / 128);
    k_gemm  <<<grid, 256, 0, stream>>>(qx, q_weight, act, w16, xs, scale_col, bias, out);
}
